// Backbone3d_4681514353314
// MI455X (gfx1250) — matrix operand order repaired and checked by reading, not compiled, not run on hardware
//
#include <hip/hip_runtime.h>
#include <hip/hip_bf16.h>
#include <cstdint>
#include <cstddef>

// ---------------------------------------------------------------------------
// CDNA5 (gfx1250) implicit-GEMM dense equivalent of the sparse 3D backbone.
// Wave32; each wave computes a 16(spatial) x 16(Cout) tile with
// v_wmma_f32_16x16x32_f16 over 27 taps x (CIN/32) K-chunks (unrolled).
// Activations: f16 channels-last [D,H,W,C] -> A fragments are b128 loads.
// Weights: pre-packed into the exact WMMA B-register layout
//   [tap][kchunk][ntile][lane][16 halves], staged per-block into LDS with
//   global_load_async_to_lds_b64 (ASYNCcnt) and read back as ds_load_b128.
// ---------------------------------------------------------------------------

typedef __attribute__((ext_vector_type(16))) _Float16 v16h;
typedef __attribute__((ext_vector_type(8)))  _Float16 v8h;
typedef __attribute__((ext_vector_type(4)))  _Float16 v4h;
typedef __attribute__((ext_vector_type(8)))  float    v8f;

#define WAVES_PER_BLOCK 8
#define BLOCK_THREADS   256

// fp32 NCDHW (N=1) -> f16 channels-last, masked by occupancy
__global__ void cvt_input_kernel(const float* __restrict__ x,
                                 const unsigned char* __restrict__ occ,
                                 _Float16* __restrict__ xout, int P, int Cin) {
  int p = blockIdx.x * blockDim.x + threadIdx.x;
  if (p >= P) return;
  float m = occ[p] ? 1.f : 0.f;
  for (int c = 0; c < Cin; ++c)
    xout[(size_t)p * Cin + c] = (_Float16)(x[(size_t)c * P + p] * m);
}

// Pack w[co][ci][t] (OIDHW fp32) into WMMA B-fragment layout (f16):
//   wq[((t*nkc + kci)*ntiles + nt)*512 + lane*16 + h]
//   with lane = (g<<4)|n : K = (h<8?h:h+8) + 8g, ci = kci*32 + K, co = nt*16 + n
__global__ void pack_w_kernel(const float* __restrict__ w,
                              _Float16* __restrict__ wq,
                              int T, int Cin, int Cout, int nkc, int ntiles) {
  int j = blockIdx.x * blockDim.x + threadIdx.x;
  int total = T * nkc * ntiles * 512;
  if (j >= total) return;
  int h512 = j & 511;
  int frag = j >> 9;
  int nt  = frag % ntiles; frag /= ntiles;
  int kci = frag % nkc;
  int t   = frag / nkc;
  int lane = h512 >> 4;
  int h    = h512 & 15;
  int g    = lane >> 4;
  int n    = lane & 15;
  int ci   = kci * 32 + (h < 8 ? h : h + 8) + 8 * g;
  int co   = nt * 16 + n;
  float v = 0.f;
  if (ci < Cin) v = w[((size_t)co * Cin + ci) * T + t];
  wq[j] = (_Float16)v;
}

// Fold conv bias into the BN affine: y = conv*s + (beta + bias*s)
__global__ void fuse_affine_kernel(const float* __restrict__ s,
                                   const float* __restrict__ beta,
                                   const float* __restrict__ bias,
                                   float* __restrict__ scale,
                                   float* __restrict__ shift, int C) {
  int c = blockIdx.x * blockDim.x + threadIdx.x;
  if (c >= C) return;
  scale[c] = s[c];
  shift[c] = beta[c] + (bias ? bias[c] * s[c] : 0.f);
}

// Strided sparse conv dilates the active set with the kernel footprint.
__global__ void dilate_occ_kernel(const unsigned char* __restrict__ oin,
                                  unsigned char* __restrict__ oout,
                                  int Din, int Hin, int Win,
                                  int Dout, int Hout, int Wout,
                                  int KD, int KH, int KW,
                                  int sd, int sh, int sw,
                                  int pd, int ph, int pw) {
  int p = blockIdx.x * blockDim.x + threadIdx.x;
  int P = Dout * Hout * Wout;
  if (p >= P) return;
  int od = p / (Hout * Wout);
  int r  = p % (Hout * Wout);
  int oh = r / Wout, ow = r % Wout;
  unsigned char v = 0;
  for (int kd = 0; kd < KD; ++kd) {
    int id = od * sd - pd + kd;
    if (id < 0 || id >= Din) continue;
    for (int kh = 0; kh < KH; ++kh) {
      int ih = oh * sh - ph + kh;
      if (ih < 0 || ih >= Hin) continue;
      for (int kw = 0; kw < KW; ++kw) {
        int iw = ow * sw - pw + kw;
        if (iw < 0 || iw >= Win) continue;
        v |= oin[((size_t)id * Hin + ih) * Win + iw];
      }
    }
  }
  oout[p] = v ? 1 : 0;
}

// Implicit-GEMM conv + fused BN/ReLU/occupancy-mask epilogue.
// grid.x = pblocks * ntiles; blockIdx -> (pblock, ntile); all waves in a
// block share ntile so the packed B fragments are staged once into LDS.
// CIN is compile-time: K-chunk loop unrolls, A-load path specializes.
template <int CIN>
__global__ __launch_bounds__(BLOCK_THREADS)
void conv_wmma_kernel(const _Float16* __restrict__ xin,
                      const _Float16* __restrict__ wq,
                      const float* __restrict__ scale,
                      const float* __restrict__ shift,
                      const unsigned char* __restrict__ occ,
                      _Float16* __restrict__ yout,
                      int Din, int Hin, int Win,
                      int Dout, int Hout, int Wout, int Cout,
                      int KD, int KH, int KW,
                      int sd, int sh, int sw,
                      int pd, int ph, int pw,
                      int ptiles, int ntiles) {
  constexpr int NKC = (CIN + 31) / 32;
  extern __shared__ _Float16 smem[];   // [T*NKC][512] B fragments (this ntile)

  int ntile  = blockIdx.x % ntiles;
  int pblock = blockIdx.x / ntiles;
  int wave   = threadIdx.x >> 5;
  int lane   = threadIdx.x & 31;
  int g      = lane >> 4;                // half-wave group
  int mn     = lane & 15;                // A/C row or B/D column

  // ---- async stage of packed B fragments into LDS (8B per lane) ----
  int T = KD * KH * KW;
  int nfrag = T * NKC;                   // fragments of 512 halves each
  {
    const uint64_t* wq64 = (const uint64_t*)wq;
    uint32_t lds_base = (uint32_t)(uintptr_t)smem;  // LDS byte offset
    int n64 = nfrag * 64;                // 64 x 8B per fragment
    for (int i = threadIdx.x; i < n64; i += BLOCK_THREADS) {
      int frag = i >> 6;
      int off  = i & 63;
      uint64_t gaddr = (uint64_t)(uintptr_t)(wq64 + ((size_t)frag * ntiles + ntile) * 64 + off);
      uint32_t laddr = lds_base + (uint32_t)i * 8u;
      asm volatile("global_load_async_to_lds_b64 %0, %1, off"
                   :: "v"(laddr), "v"(gaddr) : "memory");
    }
    asm volatile("s_wait_asynccnt 0x0" ::: "memory");
  }
  __syncthreads();

  int ptile = pblock * WAVES_PER_BLOCK + wave;
  if (ptile < ptiles) {                  // wave-uniform guard
    int P  = Dout * Hout * Wout;
    int pm = ptile * 16 + mn;            // this lane's A-row voxel (P % 16 == 0)
    int od = pm / (Hout * Wout);
    int r0 = pm % (Hout * Wout);
    int oh = r0 / Wout;
    int ow = r0 % Wout;
    int co = ntile * 16 + mn;            // this lane's B/D column channel

    v8f acc = {};

    for (int kd = 0; kd < KD; ++kd) {
      int id = od * sd - pd + kd;
      bool dok = (id >= 0) && (id < Din);
      for (int kh = 0; kh < KH; ++kh) {
        int ih = oh * sh - ph + kh;
        bool hok = dok && (ih >= 0) && (ih < Hin);
        for (int kw = 0; kw < KW; ++kw) {
          int iw = ow * sw - pw + kw;
          bool ok = hok && (iw >= 0) && (iw < Win);
          size_t abase = ok ? (((size_t)id * Hin + ih) * Win + iw) * (size_t)CIN : 0;
          int t = (kd * KH + kh) * KW + kw;

#pragma unroll
          for (int kci = 0; kci < NKC; ++kci) {
            const int kc = kci * 32;
            // ---- A fragment (layout: K = (h<8?h:h+8) + 8g + kc) ----
            v16h a = {};
            if (ok) {
              if (CIN >= 32) {           // full 32-wide chunk
                v8h lo = *(const v8h*)(xin + abase + kc + 8 * g);
                v8h hi = *(const v8h*)(xin + abase + kc + 16 + 8 * g);
#pragma unroll
                for (int h = 0; h < 8; ++h) { a[h] = lo[h]; a[h + 8] = hi[h]; }
              } else if (CIN == 16) {    // low half only
                v8h lo = *(const v8h*)(xin + abase + 8 * g);
#pragma unroll
                for (int h = 0; h < 8; ++h) a[h] = lo[h];
              } else {                   // CIN == 4: only g==0 lanes, 4 halves
                if (g == 0) {
                  v4h q = *(const v4h*)(xin + abase);
#pragma unroll
                  for (int h = 0; h < 4; ++h) a[h] = q[h];
                }
              }
            }
            // ---- B fragment: 32 contiguous bytes from LDS ----
            const _Float16* bp = &smem[((size_t)(t * NKC + kci) << 9) + lane * 16];
            v8h b0 = *(const v8h*)bp;
            v8h b1 = *(const v8h*)(bp + 8);
            v16h b;
#pragma unroll
            for (int h = 0; h < 8; ++h) { b[h] = b0[h]; b[h + 8] = b1[h]; }

            // reconverged: EXEC all-ones across the wave
            acc = __builtin_amdgcn_wmma_f32_16x16x32_f16(
                false, a, false, b, (short)0, acc, false, false);
          }
        }
      }
    }

    float sc = scale[co];
    float sf = shift[co];
    // 8 occupancy bytes for this lane's M-rows in one aligned load
    uint64_t occ8 = *(const uint64_t*)(occ + (size_t)ptile * 16 + 8 * g);
#pragma unroll
    for (int r = 0; r < 8; ++r) {
      int M = r + 8 * g;
      int p = ptile * 16 + M;
      float v = acc[r] * sc + sf;
      v = v > 0.f ? v : 0.f;             // ReLU
      if (!((occ8 >> (8 * r)) & 0xff)) v = 0.f;  // mask to active set
      yout[(size_t)p * Cout + co] = (_Float16)v;
    }
  }
}

// [D,H,W,C] f16 -> fp32 BEV map [(C*D), H, W]
__global__ void final_out_kernel(const _Float16* __restrict__ y,
                                 float* __restrict__ out,
                                 int D, int H, int W, int C) {
  int i = blockIdx.x * blockDim.x + threadIdx.x;
  int total = D * H * W * C;
  if (i >= total) return;
  int c = i % C;
  int p = i / C;
  int d = p / (H * W);
  int r = p % (H * W);
  int h = r / W, w = r % W;
  out[(((size_t)c * D + d) * H + h) * W + w] = (float)y[i];
}

// Host-side templated launcher
template <int CIN>
static void launch_conv_t(const _Float16* xin, const _Float16* wq,
                          const float* SC, const float* SH,
                          const unsigned char* occm, _Float16* yout,
                          int Din, int Hin, int Win,
                          int Dout, int Hout, int Wout, int Cout,
                          int KD, int KH, int KW, int sd, int sh, int sw,
                          int pd, int ph, int pw, hipStream_t stream) {
  constexpr int NKC = (CIN + 31) / 32;
  int P = Dout * Hout * Wout;
  int ntiles = Cout / 16;
  int ptiles = (P + 15) / 16;
  int pblocks = (ptiles + WAVES_PER_BLOCK - 1) / WAVES_PER_BLOCK;
  size_t shmem = (size_t)(KD * KH * KW) * NKC * 512 * sizeof(_Float16);
  conv_wmma_kernel<CIN><<<pblocks * ntiles, BLOCK_THREADS, shmem, stream>>>(
      xin, wq, SC, SH, occm, yout,
      Din, Hin, Win, Dout, Hout, Wout, Cout,
      KD, KH, KW, sd, sh, sw, pd, ph, pw, ptiles, ntiles);
}

extern "C" void kernel_launch(void* const* d_in, const int* in_sizes, int n_in,
                              void* d_out, int out_size, void* d_ws, size_t ws_size,
                              hipStream_t stream) {
  (void)in_sizes; (void)n_in; (void)out_size; (void)ws_size;

  // ----- inputs (setup_inputs() dict order) -----
  const float*         x    = (const float*)d_in[0];
  const unsigned char* occ0 = (const unsigned char*)d_in[1];
  const float* w_in = (const float*)d_in[2];
  const float* s_in = (const float*)d_in[3];
  const float* b_in = (const float*)d_in[4];
  const float* w1   = (const float*)d_in[5];
  const float* s1   = (const float*)d_in[6];
  const float* b1   = (const float*)d_in[7];
  const float* w2d    = (const float*)d_in[8];
  const float* bias2d = (const float*)d_in[9];
  const float* s2d    = (const float*)d_in[10];
  const float* beta2d = (const float*)d_in[11];
  const float* w2s    = (const float*)d_in[12];
  const float* s2s    = (const float*)d_in[13];
  const float* beta2s = (const float*)d_in[14];
  const float* w3d    = (const float*)d_in[15];
  const float* bias3d = (const float*)d_in[16];
  const float* s3d    = (const float*)d_in[17];
  const float* beta3d = (const float*)d_in[18];
  const float* w3s    = (const float*)d_in[19];
  const float* s3s    = (const float*)d_in[20];
  const float* beta3s = (const float*)d_in[21];
  const float* w4d    = (const float*)d_in[22];
  const float* bias4d = (const float*)d_in[23];
  const float* s4d    = (const float*)d_in[24];
  const float* beta4d = (const float*)d_in[25];
  const float* w4s    = (const float*)d_in[26];
  const float* s4s    = (const float*)d_in[27];
  const float* beta4s = (const float*)d_in[28];
  const float* wout    = (const float*)d_in[29];
  const float* biasout = (const float*)d_in[30];
  const float* sout    = (const float*)d_in[31];
  const float* betaout = (const float*)d_in[32];
  float* out = (float*)d_out;

  // ----- spatial dims -----
  const int D0 = 41, H0 = 320, W0 = 280;  const int P0 = D0 * H0 * W0;
  const int D1 = 21, H1 = 160, W1 = 140;  const int P1 = D1 * H1 * W1;
  const int D2 = 11, H2 = 80,  W2 = 70;   const int P2 = D2 * H2 * W2;
  const int D3 = 6,  H3 = 40,  W3 = 35;   const int P3 = D3 * H3 * W3;
  const int D4 = 2,  H4 = 40,  W4 = 35;   const int P4 = D4 * H4 * W4;

  // ----- workspace carve-up -----
  char* ws = (char*)d_ws;
  size_t off = 0;
  auto take = [&](size_t bytes) -> char* {
    char* p = ws + off;
    off += (bytes + 255) & ~(size_t)255;
    return p;
  };
  unsigned char* occ1 = (unsigned char*)take(P1);
  unsigned char* occ2 = (unsigned char*)take(P2);
  unsigned char* occ3 = (unsigned char*)take(P3);
  unsigned char* occ4 = (unsigned char*)take(P4);
  float*     SC = (float*)take(256 * sizeof(float));
  float*     SH = (float*)take(256 * sizeof(float));
  _Float16*  WQ = (_Float16*)take((size_t)2 << 20);              // max 27*4*8*512*2B
  _Float16*  XA = (_Float16*)take((size_t)P0 * 16 * sizeof(_Float16));
  _Float16*  XB = (_Float16*)take((size_t)P0 * 16 * sizeof(_Float16));

  // ----- per-layer helpers -----
  auto prep_w = [&](const float* w, int T, int Ci, int Co) {
    int nkc = (Ci + 31) / 32;
    int ntiles = Co / 16;
    int total = T * nkc * ntiles * 512;
    pack_w_kernel<<<(total + 255) / 256, 256, 0, stream>>>(w, WQ, T, Ci, Co, nkc, ntiles);
  };
  auto prep_aff = [&](const float* s, const float* beta, const float* bias, int C) {
    fuse_affine_kernel<<<1, 256, 0, stream>>>(s, beta, bias, SC, SH, C);
  };
  auto conv = [&](const _Float16* xin, _Float16* yout,
                  int Din, int Hin, int Win, int Cin,
                  int Dout, int Hout, int Wout, int Cout,
                  int KD, int KH, int KW, int sd, int sh, int sw,
                  int pd, int ph, int pw, const unsigned char* occm) {
    switch (Cin) {
      case 4:
        launch_conv_t<4>(xin, WQ, SC, SH, occm, yout, Din, Hin, Win,
                         Dout, Hout, Wout, Cout, KD, KH, KW, sd, sh, sw,
                         pd, ph, pw, stream);
        break;
      case 16:
        launch_conv_t<16>(xin, WQ, SC, SH, occm, yout, Din, Hin, Win,
                          Dout, Hout, Wout, Cout, KD, KH, KW, sd, sh, sw,
                          pd, ph, pw, stream);
        break;
      case 32:
        launch_conv_t<32>(xin, WQ, SC, SH, occm, yout, Din, Hin, Win,
                          Dout, Hout, Wout, Cout, KD, KH, KW, sd, sh, sw,
                          pd, ph, pw, stream);
        break;
      case 64:
        launch_conv_t<64>(xin, WQ, SC, SH, occm, yout, Din, Hin, Win,
                          Dout, Hout, Wout, Cout, KD, KH, KW, sd, sh, sw,
                          pd, ph, pw, stream);
        break;
      default:
        launch_conv_t<128>(xin, WQ, SC, SH, occm, yout, Din, Hin, Win,
                           Dout, Hout, Wout, Cout, KD, KH, KW, sd, sh, sw,
                           pd, ph, pw, stream);
        break;
    }
  };

  // ----- layer chain (in-order stream serializes WQ/SC/SH reuse) -----
  cvt_input_kernel<<<(P0 + 255) / 256, 256, 0, stream>>>(x, occ0, XA, P0, 4);

  // conv_input: SubM 4->16
  prep_w(w_in, 27, 4, 16);  prep_aff(s_in, b_in, nullptr, 16);
  conv(XA, XB, D0,H0,W0,4,  D0,H0,W0,16, 3,3,3, 1,1,1, 1,1,1, occ0);

  // conv1: SubM 16->16
  prep_w(w1, 27, 16, 16);   prep_aff(s1, b1, nullptr, 16);
  conv(XB, XA, D0,H0,W0,16, D0,H0,W0,16, 3,3,3, 1,1,1, 1,1,1, occ0);

  // conv2 down: 16->32, stride 2
  dilate_occ_kernel<<<(P1 + 255) / 256, 256, 0, stream>>>(
      occ0, occ1, D0,H0,W0, D1,H1,W1, 3,3,3, 2,2,2, 1,1,1);
  prep_w(w2d, 27, 16, 32);  prep_aff(s2d, beta2d, bias2d, 32);
  conv(XA, XB, D0,H0,W0,16, D1,H1,W1,32, 3,3,3, 2,2,2, 1,1,1, occ1);

  // 3x SubM 32->32  (XB -> XA -> XB -> XA)
  {
    _Float16 *cur = XB, *oth = XA;
    for (int i = 0; i < 3; ++i) {
      prep_w(w2s + (size_t)i * 32 * 32 * 27, 27, 32, 32);
      prep_aff(s2s + i * 32, beta2s + i * 32, nullptr, 32);
      conv(cur, oth, D1,H1,W1,32, D1,H1,W1,32, 3,3,3, 1,1,1, 1,1,1, occ1);
      _Float16* t = cur; cur = oth; oth = t;
    }
  }

  // conv3 down: 32->64, stride 2
  dilate_occ_kernel<<<(P2 + 255) / 256, 256, 0, stream>>>(
      occ1, occ2, D1,H1,W1, D2,H2,W2, 3,3,3, 2,2,2, 1,1,1);
  prep_w(w3d, 27, 32, 64);  prep_aff(s3d, beta3d, bias3d, 64);
  conv(XA, XB, D1,H1,W1,32, D2,H2,W2,64, 3,3,3, 2,2,2, 1,1,1, occ2);

  // 3x SubM 64->64
  {
    _Float16 *cur = XB, *oth = XA;
    for (int i = 0; i < 3; ++i) {
      prep_w(w3s + (size_t)i * 64 * 64 * 27, 27, 64, 64);
      prep_aff(s3s + i * 64, beta3s + i * 64, nullptr, 64);
      conv(cur, oth, D2,H2,W2,64, D2,H2,W2,64, 3,3,3, 1,1,1, 1,1,1, occ2);
      _Float16* t = cur; cur = oth; oth = t;
    }
  }

  // conv4 down: 64->128, stride 2
  dilate_occ_kernel<<<(P3 + 255) / 256, 256, 0, stream>>>(
      occ2, occ3, D2,H2,W2, D3,H3,W3, 3,3,3, 2,2,2, 1,1,1);
  prep_w(w4d, 27, 64, 128); prep_aff(s4d, beta4d, bias4d, 128);
  conv(XA, XB, D2,H2,W2,64, D3,H3,W3,128, 3,3,3, 2,2,2, 1,1,1, occ3);

  // 3x SubM 128->128
  {
    _Float16 *cur = XB, *oth = XA;
    for (int i = 0; i < 3; ++i) {
      prep_w(w4s + (size_t)i * 128 * 128 * 27, 27, 128, 128);
      prep_aff(s4s + i * 128, beta4s + i * 128, nullptr, 128);
      conv(cur, oth, D3,H3,W3,128, D3,H3,W3,128, 3,3,3, 1,1,1, 1,1,1, occ3);
      _Float16* t = cur; cur = oth; oth = t;
    }
  }

  // conv_out: 128->256, k=(3,1,1), stride=(2,1,1), pad 0
  dilate_occ_kernel<<<(P4 + 255) / 256, 256, 0, stream>>>(
      occ3, occ4, D3,H3,W3, D4,H4,W4, 3,1,1, 2,1,1, 0,0,0);
  prep_w(wout, 3, 128, 256); prep_aff(sout, betaout, biasout, 256);
  conv(XA, XB, D3,H3,W3,128, D4,H4,W4,256, 3,1,1, 2,1,1, 0,0,0, occ4);

  // reshape to dense BEV map, fp32
  final_out_kernel<<<(P4 * 256 + 255) / 256, 256, 0, stream>>>(XB, out, D4, H4, W4, 256);
}
